// RecurrentNet_77025943486685
// MI455X (gfx1250) — matrix operand order repaired and checked by reading, not compiled, not run on hardware
//
#include <hip/hip_runtime.h>
#include <hip/hip_bf16.h>
#include <math.h>

typedef __attribute__((ext_vector_type(16))) _Float16 v16h;
typedef __attribute__((ext_vector_type(8)))  _Float16 v8h;
typedef __attribute__((ext_vector_type(8)))  float    v8f;

#define BT       16     // batch rows per workgroup
#define U1N      128    // GRU-1 units
#define G1N      384    // 3*U1
#define U2N      32     // GRU-2 units
#define G2N      96     // 3*U2
#define NTHREADS 256    // 8 waves (wave32)

// Fast transcendental paths: gfx1250 has V_TANH_F32 / V_RCP_F32 / V_EXP_F32.
#if __has_builtin(__builtin_amdgcn_tanhf)
__device__ __forceinline__ float fast_tanh(float x) { return __builtin_amdgcn_tanhf(x); }
#else
__device__ __forceinline__ float fast_tanh(float x) { return tanhf(x); }
#endif

#if __has_builtin(__builtin_amdgcn_rcpf)
__device__ __forceinline__ float fast_rcp(float x) { return __builtin_amdgcn_rcpf(x); }
#else
__device__ __forceinline__ float fast_rcp(float x) { return 1.0f / x; }
#endif

__device__ __forceinline__ float fast_sigmoid(float x) {
  return fast_rcp(1.0f + __expf(-x));   // v_exp_f32 + v_rcp_f32
}

union HFrag { v16h v; v8h h[2]; };

// A-fragment (16x32 f16, M x K) from row-major LDS. Per ISA 7.12.2:
// lanes 0-15 hold row M=lane, K in {0..7, 16..23}; lanes 16-31 hold row
// M=lane-16, K in {8..15, 24..31}. Two contiguous 16B chunks per lane.
__device__ __forceinline__ v16h load_a_frag_lds(const _Float16* base, int ld, int ktile) {
  const int lane = threadIdx.x & 31;
  const int row  = lane & 15;
  const int hi   = lane >> 4;
  const _Float16* p = base + row * ld + ktile * 32 + hi * 8;
  HFrag f;
  f.h[0] = *(const v8h*)(p);        // K: ktile*32 + hi*8 + 0..7
  f.h[1] = *(const v8h*)(p + 16);   // K: ktile*32 + hi*8 + 16..23
  return f.v;
}

// B-fragment (32x16 f16, K x N) gathered from row-major f32 global weights.
// Lane L holds column N=L%16; lanes 0-15 K={0..7,16..23}, lanes 16-31 K={8..15,24..31}.
// Init-only cost; fragments stay VGPR-resident for all T steps.
__device__ __forceinline__ v16h load_b_frag_gmem(const float* g, int ld, int kbase, int colbase) {
  const int lane = threadIdx.x & 31;
  const int c  = colbase + (lane & 15);
  const int kb = kbase + (lane >> 4) * 8;
  v16h b;
  #pragma unroll
  for (int i = 0; i < 16; ++i) b[i] = (_Float16)g[(size_t)(kb + (i & 7) + (i >> 3) * 16) * ld + c];
  return b;
}

// C/D fragment (16x16 f32): VGPR r, lane L -> M = r + 8*(L/16), N = L%16.
__device__ __forceinline__ void store_d_lds(float* base, int ld, int colbase, v8f d) {
  const int lane = threadIdx.x & 31;
  const int c  = colbase + (lane & 15);
  const int r0 = (lane >> 4) * 8;
  #pragma unroll
  for (int r = 0; r < 8; ++r) base[(r0 + r) * ld + c] = d[r];
}

__device__ __forceinline__ v8f wmma_f16f32(v16h a, v16h b, v8f c) {
  // (neg_a, A, neg_b, B, c_mod, C, reuse_a, reuse_b)
  return __builtin_amdgcn_wmma_f32_16x16x32_f16(false, a, false, b, (short)0, c, false, false);
}

__global__ __launch_bounds__(NTHREADS, 1)
void gru2_fused_kernel(const float* __restrict__ x,    // [B,T,1]
                       const float* __restrict__ k1,   // [1,384]
                       const float* __restrict__ rk1,  // [128,384]
                       const float* __restrict__ b1,   // [2,384]
                       const float* __restrict__ k2,   // [128,96]
                       const float* __restrict__ rk2,  // [32,96]
                       const float* __restrict__ b2,   // [2,96]
                       const float* __restrict__ wd,   // [32,1]
                       const float* __restrict__ bd,   // [1]
                       float* __restrict__ out,        // [B,1]
                       int T)
{
  __shared__ float    s_h1 [BT][U1N];   // GRU-1 state fp32
  __shared__ _Float16 s_h1h[BT][U1N];   // GRU-1 state f16 (WMMA A operand)
  __shared__ float    s_in1[BT][G1N];   // h1 @ rk1 per step
  __shared__ float    s_x2 [BT][G2N];   // h1_new @ k2
  __shared__ float    s_i2 [BT][G2N];   // h2 @ rk2
  __shared__ float    s_h2 [BT][U2N];
  __shared__ _Float16 s_h2h[BT][U2N];
  __shared__ float    s_k1 [G1N];
  __shared__ float    s_b1i[G1N];
  __shared__ float    s_b1r[G1N];
  __shared__ float    s_b2i[G2N];
  __shared__ float    s_b2r[G2N];
  __shared__ float    s_xt [BT];        // x values for current step

  const int tid  = threadIdx.x;
  const int lane = tid & 31;
  const int wave = tid >> 5;
  const int b0   = blockIdx.x * BT;
  const int nt0  = wave * 3;            // 3 N-tiles of rk1 per wave (24 total)

  // ---- init: weights -> per-wave register-resident WMMA B fragments ----
  v16h brk1[3][4];                      // [ntile][ktile]  (96 VGPRs, resident all T)
  #pragma unroll
  for (int j = 0; j < 3; ++j)
    #pragma unroll
    for (int k = 0; k < 4; ++k)
      brk1[j][k] = load_b_frag_gmem(rk1, G1N, k * 32, (nt0 + j) * 16);

  v16h bk2[4] = {}; v16h brk2 = {};
  if (wave < 6) {                       // 6 N-tiles cover G2N=96
    #pragma unroll
    for (int k = 0; k < 4; ++k) bk2[k] = load_b_frag_gmem(k2, G2N, k * 32, wave * 16);
    brk2 = load_b_frag_gmem(rk2, G2N, 0, wave * 16);
  }

  for (int i = tid; i < G1N; i += NTHREADS) {
    s_k1 [i] = k1[i];
    s_b1i[i] = b1[i];
    s_b1r[i] = b1[G1N + i];
  }
  for (int i = tid; i < G2N; i += NTHREADS) {
    s_b2i[i] = b2[i];
    s_b2r[i] = b2[G2N + i];
  }
  for (int i = tid; i < BT * U1N; i += NTHREADS) {
    (&s_h1[0][0])[i]  = 0.0f;
    (&s_h1h[0][0])[i] = (_Float16)0.0f;
  }
  for (int i = tid; i < BT * U2N; i += NTHREADS) {
    (&s_h2[0][0])[i]  = 0.0f;
    (&s_h2h[0][0])[i] = (_Float16)0.0f;
  }
  // x[b,t] software pipeline lives in wave 7 (idle during phase-C WMMAs).
  float xnext = 0.0f;
  if (wave == 7 && lane < BT) {
    s_xt[lane] = x[(size_t)(b0 + lane) * T];
    xnext      = x[(size_t)(b0 + lane) * T + (T > 1 ? 1 : 0)];
  }
  __syncthreads();

  for (int t = 0; t < T; ++t) {
    // ---- Phase A: inner1 = h1 @ rk1  (12 WMMAs / wave) ----
    {
      v16h a0 = load_a_frag_lds(&s_h1h[0][0], U1N, 0);
      v16h a1 = load_a_frag_lds(&s_h1h[0][0], U1N, 1);
      v16h a2 = load_a_frag_lds(&s_h1h[0][0], U1N, 2);
      v16h a3 = load_a_frag_lds(&s_h1h[0][0], U1N, 3);
      #pragma unroll
      for (int j = 0; j < 3; ++j) {
        v8f acc = {};
        acc = wmma_f16f32(a0, brk1[j][0], acc);
        acc = wmma_f16f32(a1, brk1[j][1], acc);
        acc = wmma_f16f32(a2, brk1[j][2], acc);
        acc = wmma_f16f32(a3, brk1[j][3], acc);
        store_d_lds(&s_in1[0][0], G1N, (nt0 + j) * 16, acc);
      }
    }
    __syncthreads();

    // ---- Phase B: GRU-1 gates (z,r: relu; act: tanh), 16x128 elementwise ----
    #pragma unroll
    for (int i = 0; i < (BT * U1N) / NTHREADS; ++i) {
      int idx = tid + i * NTHREADS;
      int row = idx >> 7;
      int u   = idx & (U1N - 1);
      float xv = s_xt[row];                      // F==1: input proj = x * k1 row
      float xz = fmaf(xv, s_k1[u],           s_b1i[u]);
      float xr = fmaf(xv, s_k1[U1N + u],     s_b1i[U1N + u]);
      float xh = fmaf(xv, s_k1[2*U1N + u],   s_b1i[2*U1N + u]);
      float iz = s_in1[row][u]         + s_b1r[u];
      float ir = s_in1[row][U1N + u]   + s_b1r[U1N + u];
      float ih = s_in1[row][2*U1N + u] + s_b1r[2*U1N + u];
      float z  = fmaxf(xz + iz, 0.0f);
      float r  = fmaxf(xr + ir, 0.0f);
      float hh = fast_tanh(xh + r * ih);         // V_TANH_F32 (TRANS), branch-free
      float h  = s_h1[row][u];
      float hn = z * h + (1.0f - z) * hh;
      s_h1 [row][u] = hn;
      s_h1h[row][u] = (_Float16)hn;
    }
    __syncthreads();

    // ---- Phase C: x2 = h1_new @ k2 ; i2 = h2 @ rk2 (waves 0..5, 5 WMMAs) ----
    if (wave < 6) {
      v16h c0 = load_a_frag_lds(&s_h1h[0][0], U1N, 0);
      v16h c1 = load_a_frag_lds(&s_h1h[0][0], U1N, 1);
      v16h c2 = load_a_frag_lds(&s_h1h[0][0], U1N, 2);
      v16h c3 = load_a_frag_lds(&s_h1h[0][0], U1N, 3);
      v8f ax = {};
      ax = wmma_f16f32(c0, bk2[0], ax);
      ax = wmma_f16f32(c1, bk2[1], ax);
      ax = wmma_f16f32(c2, bk2[2], ax);
      ax = wmma_f16f32(c3, bk2[3], ax);
      store_d_lds(&s_x2[0][0], G2N, wave * 16, ax);
      v16h h2a = load_a_frag_lds(&s_h2h[0][0], U2N, 0);
      v8f ai = {};
      ai = wmma_f16f32(h2a, brk2, ai);
      store_d_lds(&s_i2[0][0], G2N, wave * 16, ai);
    } else if (wave == 7 && lane < BT) {
      // advance pipelined x load; overlaps the WMMA phase on waves 0..5
      s_xt[lane] = xnext;
      int tn = (t + 2 < T) ? (t + 2) : (T - 1);
      xnext = x[(size_t)(b0 + lane) * T + tn];
    }
    __syncthreads();

    // ---- Phase D: GRU-2 gates (z,r: sigmoid; act: relu), 16x32 elementwise ----
    #pragma unroll
    for (int i = 0; i < (BT * U2N) / NTHREADS; ++i) {
      int idx = tid + i * NTHREADS;
      int row = idx >> 5;
      int u   = idx & (U2N - 1);
      float xz = s_x2[row][u]         + s_b2i[u];
      float xr = s_x2[row][U2N + u]   + s_b2i[U2N + u];
      float xh = s_x2[row][2*U2N + u] + s_b2i[2*U2N + u];
      float iz = s_i2[row][u]         + s_b2r[u];
      float ir = s_i2[row][U2N + u]   + s_b2r[U2N + u];
      float ih = s_i2[row][2*U2N + u] + s_b2r[2*U2N + u];
      float z  = fast_sigmoid(xz + iz);
      float r  = fast_sigmoid(xr + ir);
      float hh = fmaxf(xh + r * ih, 0.0f);
      float h  = s_h2[row][u];
      float hn = z * h + (1.0f - z) * hh;
      s_h2 [row][u] = hn;
      s_h2h[row][u] = (_Float16)hn;
    }
    __syncthreads();
  }

  // ---- dense head: out[b] = h2[b,:] . wd + bd ----
  if (tid < BT) {
    float acc = bd[0];
    #pragma unroll
    for (int u = 0; u < U2N; ++u) acc = fmaf(s_h2[tid][u], wd[u], acc);
    out[b0 + tid] = acc;
  }
}

extern "C" void kernel_launch(void* const* d_in, const int* in_sizes, int n_in,
                              void* d_out, int out_size, void* d_ws, size_t ws_size,
                              hipStream_t stream) {
  const float* x   = (const float*)d_in[0];
  const float* k1  = (const float*)d_in[1];
  const float* rk1 = (const float*)d_in[2];
  const float* b1  = (const float*)d_in[3];
  const float* k2  = (const float*)d_in[4];
  const float* rk2 = (const float*)d_in[5];
  const float* b2  = (const float*)d_in[6];
  const float* wd  = (const float*)d_in[7];
  const float* bd  = (const float*)d_in[8];
  float* out = (float*)d_out;

  const int B = out_size;             // output is [B,1]
  const int T = in_sizes[0] / B;      // x is [B,T,1]

  dim3 grid(B / BT), block(NTHREADS);
  gru2_fused_kernel<<<grid, block, 0, stream>>>(x, k1, rk1, b1, k2, rk2, b2, wd, bd, out, T);
}
